// BinarizeLinear_50886772523465
// MI455X (gfx1250) — matrix operand order repaired and checked by reading, not compiled, not run on hardware
//
#include <hip/hip_runtime.h>

// ---------------------------------------------------------------------------
// BinarizeLinear: out[n,o] = sum_i x[n,i]*sign(w[o,i]) + bias[o]
// N=262144, I=O=512, fp32.
//
// Roofline: 137 GFLOP vs ~1.07 GB HBM traffic (~46 us @ 23.3 TB/s) -> keep
// it memory bound by using bf16 WMMA. sign(w) is exact in bf16; x is split
// x = hi + lo (two bf16) with f32 accumulation -> ~fp32-faithful at 2x WMMA
// cost (274 GFLOP bf16 WMMA, still near the bandwidth floor).
// ---------------------------------------------------------------------------

typedef __attribute__((ext_vector_type(16))) __bf16 v16bf;
typedef __attribute__((ext_vector_type(8)))  __bf16 v8bf;
typedef __attribute__((ext_vector_type(4)))  __bf16 v4bf;
typedef __attribute__((ext_vector_type(8)))  float  v8f;
typedef __attribute__((ext_vector_type(4)))  float  v4f;

#define KDIM  512
#define ODIM  512
#define NROWS 262144
#define BM 128
#define BN 128
#define BK 32
#define LDT 40   // LDS row stride in bf16 elements (32 data + 8 pad = 80 B)

static __device__ __forceinline__ v16bf cat16(v8bf a, v8bf b) {
  return __builtin_shufflevector(a, b, 0,1,2,3,4,5,6,7,8,9,10,11,12,13,14,15);
}

__global__ __launch_bounds__(256) void binlin_sign_bf16(
    const float* __restrict__ w, __bf16* __restrict__ sb) {
  int i = blockIdx.x * 256 + threadIdx.x;
  float v = w[i];
  float s = (v > 0.f) ? 1.f : ((v < 0.f) ? -1.f : 0.f);
  sb[i] = (__bf16)s;
}

__global__ __launch_bounds__(256) void binlin_gemm_wmma(
    const float* __restrict__ x, const __bf16* __restrict__ sb,
    const float* __restrict__ bias, float* __restrict__ out) {
  __shared__ __attribute__((aligned(16))) __bf16 Xh[BM * LDT];
  __shared__ __attribute__((aligned(16))) __bf16 Xl[BM * LDT];
  __shared__ __attribute__((aligned(16))) __bf16 Sw[BN * LDT];

  const int t    = threadIdx.x;
  const int lane = t & 31;
  const int wid  = t >> 5;
  const int wm   = wid >> 2;     // 0..1  (M direction)
  const int wn   = wid & 3;      // 0..3  (O direction)
  const int hih  = lane >> 4;    // upper half-wave
  const int l15  = lane & 15;

  const long RB = (long)blockIdx.x * BM;   // row base in x/out
  const int  OB = blockIdx.y * BN;         // col base in out / row base in sb

  v8f acc[4][2];
  const v8f vzero = {0.f, 0.f, 0.f, 0.f, 0.f, 0.f, 0.f, 0.f};
#pragma unroll
  for (int mt = 0; mt < 4; ++mt)
#pragma unroll
    for (int nt = 0; nt < 2; ++nt) acc[mt][nt] = vzero;

  for (int k0 = 0; k0 < KDIM; k0 += BK) {
    // ---- stage x tile: fp32 -> hi/lo bf16 split into LDS -----------------
#pragma unroll
    for (int i = 0; i < 4; ++i) {
      int q = i * 256 + t;                  // float4 chunk id, 1024 total
      int r = q >> 3;                       // tile row 0..127
      int c = (q & 7) * 4;                  // tile col 0..28
      const float* gp = x + (RB + r) * KDIM + k0 + c;
      v4f v = *(const v4f*)gp;
      if (k0 + BK < KDIM) __builtin_prefetch(gp + BK, 0, 1);
      __bf16 h0 = (__bf16)v.x, h1 = (__bf16)v.y, h2 = (__bf16)v.z, h3 = (__bf16)v.w;
      __bf16 l0 = (__bf16)(v.x - (float)h0);
      __bf16 l1 = (__bf16)(v.y - (float)h1);
      __bf16 l2 = (__bf16)(v.z - (float)h2);
      __bf16 l3 = (__bf16)(v.w - (float)h3);
      v4bf hv = {h0, h1, h2, h3};
      v4bf lv = {l0, l1, l2, l3};
      *(v4bf*)(&Xh[r * LDT + c]) = hv;
      *(v4bf*)(&Xl[r * LDT + c]) = lv;
    }
    // ---- stage sign(w) tile (already bf16) -------------------------------
#pragma unroll
    for (int i = 0; i < 2; ++i) {
      int p = i * 256 + t;                  // 8-bf16 chunk id, 512 total
      int r = p >> 2;
      int c = (p & 3) * 8;
      *(v8bf*)(&Sw[r * LDT + c]) = *(const v8bf*)(sb + (OB + r) * KDIM + k0 + c);
    }
    __syncthreads();

    // ---- A fragments: 16-bit A 16x32 layout (half-wave K chunks) ---------
    v16bf ah[4], al[4];
#pragma unroll
    for (int mt = 0; mt < 4; ++mt) {
      int row = wm * 64 + mt * 16 + l15;
      int cb  = hih * 8;
      ah[mt] = cat16(*(const v8bf*)(&Xh[row * LDT + cb]),
                     *(const v8bf*)(&Xh[row * LDT + cb + 16]));
      al[mt] = cat16(*(const v8bf*)(&Xl[row * LDT + cb]),
                     *(const v8bf*)(&Xl[row * LDT + cb + 16]));
    }
    // ---- B fragments: 16-bit B 32x16 (half-wave K chunks, same as A) ----
    v16bf bfr[2];
#pragma unroll
    for (int nt = 0; nt < 2; ++nt) {
      int row = wn * 32 + nt * 16 + l15;
      int cb  = hih * 8;
      bfr[nt] = cat16(*(const v8bf*)(&Sw[row * LDT + cb]),
                      *(const v8bf*)(&Sw[row * LDT + cb + 16]));
    }
#pragma unroll
    for (int mt = 0; mt < 4; ++mt)
#pragma unroll
      for (int nt = 0; nt < 2; ++nt) {
        acc[mt][nt] = __builtin_amdgcn_wmma_f32_16x16x32_bf16(
            false, ah[mt], false, bfr[nt], (short)0, acc[mt][nt], false, false);
        acc[mt][nt] = __builtin_amdgcn_wmma_f32_16x16x32_bf16(
            false, al[mt], false, bfr[nt], (short)0, acc[mt][nt], false, false);
      }
    __syncthreads();
  }

  // ---- epilogue: add bias, store (C/D layout: row = j + 8*hih, col = l15) -
#pragma unroll
  for (int nt = 0; nt < 2; ++nt) {
    int col  = OB + wn * 32 + nt * 16 + l15;
    float bv = bias[col];
#pragma unroll
    for (int mt = 0; mt < 4; ++mt) {
      long rowb = RB + wm * 64 + mt * 16 + hih * 8;
#pragma unroll
      for (int j = 0; j < 8; ++j)
        out[(rowb + j) * ODIM + col] = acc[mt][nt][j] + bv;
    }
  }
}

extern "C" void kernel_launch(void* const* d_in, const int* in_sizes, int n_in,
                              void* d_out, int out_size, void* d_ws, size_t ws_size,
                              hipStream_t stream) {
  (void)in_sizes; (void)n_in; (void)out_size; (void)ws_size;
  const float* x    = (const float*)d_in[0];
  const float* w    = (const float*)d_in[1];
  const float* bias = (const float*)d_in[2];
  float* out        = (float*)d_out;
  __bf16* sb        = (__bf16*)d_ws;   // 512*512*2 B = 512 KB sign matrix

  binlin_sign_bf16<<<(ODIM * KDIM) / 256, 256, 0, stream>>>(w, sb);
  binlin_gemm_wmma<<<dim3(NROWS / BM, ODIM / BN), 256, 0, stream>>>(x, sb, bias, out);
}
